// InfoGraphExplainer_82884278878812
// MI455X (gfx1250) — compile-verified, matrix operand order checked by reading, not run on hardware
//
#include <hip/hip_runtime.h>
#include <hip/hip_bf16.h>

typedef __attribute__((ext_vector_type(16))) __bf16 v16bf;
typedef __attribute__((ext_vector_type(8)))  float  v8f;

#define NNODE 4096
#define CCH   128
#define TT    12

__device__ __forceinline__ v8f wmma_bf16(v16bf a, v16bf b, v8f c) {
  return __builtin_amdgcn_wmma_f32_16x16x32_bf16(false, a, false, b, (short)0, c,
                                                 false, false);
}

// ---- fragment-native layout helpers (ISA 7.12.2, 16-bit A/B) ----------------
// A 16x32 tile: lane L holds row m=L&15; half hb=L>>4; element j:
//   j<8  -> kr = hb*8 + j
//   j>=8 -> kr = 16 + hb*8 + (j-8)
// Packed storage: tile kt is 512 contiguous bf16: [(kt*32+lane)*16 + j].
__device__ __forceinline__ int afrag_idx(int m, int k) {
  int kt = k >> 5, kr = k & 31;
  int g  = kr >> 4;             // 0: kr<16, 1: kr>=16
  int bs = kr - (g << 4);       // 0..15
  int hb = bs >> 3;
  int j  = (bs & 7) + (g << 3);
  return ((kt << 5) + (hb << 4) + m) * 16 + j;
}

__device__ __forceinline__ v16bf load_a_packed(const __bf16* tile, int kt, int lane) {
  return *(const v16bf*)(tile + (((size_t)kt << 5) + lane) * 16);
}

// packed weights: [(kt*ntiles + nt)*32 + lane]*16 + j
__device__ __forceinline__ v16bf load_b_packed(const __bf16* Wp, int ntiles,
                                               int kt, int nt, int lane) {
  return *(const v16bf*)(Wp + (((size_t)(kt * ntiles + nt) << 5) + lane) * 16);
}

__global__ void k_zero(float* p, int n) {
  int i = blockIdx.x * blockDim.x + threadIdx.x;
  if (i < n) p[i] = 0.f;
}

// pack f32 [K,N] weight into fragment-native bf16 tiles
__global__ void k_pack(const float* __restrict__ W, __bf16* __restrict__ dst,
                       int K, int N) {
  int total = K * N;
  for (int i = blockIdx.x * blockDim.x + threadIdx.x; i < total;
       i += gridDim.x * blockDim.x) {
    int tile = i >> 9, within = i & 511;
    int lane = within >> 4, j = within & 15;
    int ntiles = N >> 4;
    int kt = tile / ntiles, nt = tile - kt * ntiles;
    int n  = (nt << 4) + (lane & 15);
    int hb = lane >> 4;
    int g  = j >> 3;
    int kr = (g << 4) + (hb << 3) + (j & 7);
    int k  = (kt << 5) + kr;
    dst[i] = (__bf16)W[(size_t)k * N + n];
  }
}

// ---------------- Temporal attention: one workgroup per (b, n) ----------------
__global__ __launch_bounds__(128) void k_attn(
    const float* __restrict__ h_in,
    const __bf16* __restrict__ Wqp, const __bf16* __restrict__ Wkp,
    const __bf16* __restrict__ Wvp,
    const float* __restrict__ bq, const float* __restrict__ bk,
    const float* __restrict__ bv,
    float* __restrict__ emb_mean) {
  __shared__ __align__(32) __bf16 xbf[16 * CCH];  // fragment-native, 4 kt tiles
  // 16 rows (12 real + 4 pad) so D-stores need no row guard
  __shared__ float qs[16 * CCH], ks[16 * CCH], vs[16 * CCH];
  __shared__ float sc[TT * TT];
  __shared__ float wsum[TT];

  int b = blockIdx.x >> 12;      // / 4096
  int n = blockIdx.x & 4095;
  int tid = threadIdx.x;
  int lane = tid & 31, wv = tid >> 5;

  { // gather x[b,n,t,c] = h_in[b,c,t,n] into fragment-native bf16 layout
    int c = tid;
    const float* base = h_in + (((size_t)(b * CCH + c)) * TT) * NNODE + n;
#pragma unroll
    for (int t = 0; t < TT; ++t)
      xbf[afrag_idx(t, c)] = (__bf16)base[(size_t)t * NNODE];
#pragma unroll
    for (int t = TT; t < 16; ++t) xbf[afrag_idx(t, c)] = (__bf16)0.f;
  }
  __syncthreads();

  // q/k/v = x @ W + b : 3 matrices x 8 column tiles, 24 jobs over 4 waves
  for (int j = wv; j < 24; j += 4) {
    int mat = j >> 3, nt = j & 7;
    const __bf16* W   = (mat == 0) ? Wqp : (mat == 1) ? Wkp : Wvp;
    const float*  bia = (mat == 0) ? bq  : (mat == 1) ? bk  : bv;
    float bval = bia[nt * 16 + (lane & 15)];
    v8f d;
#pragma unroll
    for (int r = 0; r < 8; ++r) d[r] = bval;
#pragma unroll
    for (int kt = 0; kt < 4; ++kt) {
      v16bf a  = load_a_packed(xbf, kt, lane);
      v16bf bb = load_b_packed(W, 8, kt, nt, lane);
      d = wmma_bf16(a, bb, d);
    }
    float* outm = (mat == 0) ? qs : (mat == 1) ? ks : vs;
    float* op = outm + ((lane >> 4) * 8) * CCH + nt * 16 + (lane & 15);
#pragma unroll
    for (int r = 0; r < 8; ++r) op[r * CCH] = d[r];  // unguarded: rows 12..15 pad
  }
  __syncthreads();

  if (tid < TT * TT) {  // scores = q k^T / sqrt(C)
    int t = tid / TT, s = tid % TT;
    float acc = 0.f;
    for (int c = 0; c < CCH; ++c) acc += qs[t * CCH + c] * ks[s * CCH + c];
    sc[t * TT + s] = acc * 0.08838834764831845f;  // 1/sqrt(128)
  }
  __syncthreads();
  if (tid < TT) {  // softmax over s, per row t
    int t = tid;
    float mx = -1e30f;
    for (int s = 0; s < TT; ++s) mx = fmaxf(mx, sc[t * TT + s]);
    float e_[TT], sum = 0.f;
    for (int s = 0; s < TT; ++s) { e_[s] = __expf(sc[t * TT + s] - mx); sum += e_[s]; }
    float inv = 1.f / sum;
    for (int s = 0; s < TT; ++s) sc[t * TT + s] = e_[s] * inv;
  }
  __syncthreads();
  if (tid < TT) {  // w[s] = sum_t attn[t,s]  (emb sums over t)
    int s = tid;
    float acc = 0.f;
    for (int t = 0; t < TT; ++t) acc += sc[t * TT + s];
    wsum[s] = acc;
  }
  __syncthreads();
  {  // emb[c] = sum_s w[s] v[s,c];  emb_mean += emb / B
    int c = tid;
    float acc = 0.f;
    for (int s = 0; s < TT; ++s) acc += wsum[s] * vs[s * CCH + c];
    atomicAdd(&emb_mean[n * CCH + c], acc * 0.125f);
  }
}

// ---------------- Edge MLP: 16 edges per wave, 2 waves per block --------------
// stats layout: [0..511] sum1, [512..1023] sq1, [1024..1151] sum2,
//               [1152..1279] sq2, [1280] cnt
template <int PASS>
__global__ __launch_bounds__(64) void k_mlp(
    const float* __restrict__ emb_mean,
    const int* __restrict__ row, const int* __restrict__ col,
    const __bf16* __restrict__ W1p, const float* __restrict__ b1,
    const __bf16* __restrict__ W2p, const float* __restrict__ b2,
    const float* __restrict__ W3,  const float* __restrict__ b3,
    float* __restrict__ stats, float* __restrict__ att, int E) {
  __shared__ __align__(32) unsigned char smem[2 * 24576];
  __shared__ float maskS[2][16];
  int lane = threadIdx.x & 31, wv = threadIdx.x >> 5;
  unsigned char* base = smem + wv * 24576;
  __bf16* A1 = (__bf16*)base;           // 16x256 fragment-native (8 KB)
  __bf16* A2 = (__bf16*)(base + 8192);  // 16x512 fragment-native (16 KB)
  float*  H2 = (float*)base;            // 16x128 f32, reuses A1 region in PASS 3

  int e0 = (blockIdx.x * 2 + wv) * 16;

  if (lane < 16) {
    int e = e0 + lane;
    maskS[wv][lane] = (row[e] != col[e]) ? 1.f : 0.f;
  }
  // build f12 tile [emb_mean[row], emb_mean[col]] in fragment-native layout
  for (int idx = lane; idx < 16 * 256; idx += 32) {
    int m = idx >> 8, f = idx & 255;
    int e = e0 + m;
    int node = (f < 128) ? row[e] : col[e];
    A1[afrag_idx(m, f)] = (__bf16)emb_mean[node * CCH + (f & 127)];
  }
  __syncthreads();

  if (PASS == 1) {  // masked edge count
    unsigned long long bal = __ballot(lane < 16 && maskS[wv][lane] != 0.f);
    if (lane == 0) atomicAdd(&stats[1280], (float)__popcll(bal));
  }

  int nn = lane & 15, mb = (lane >> 4) * 8;

  // ---- layer 1: [16,256] @ [256,512] + b1 ----
  for (int nt = 0; nt < 32; ++nt) {
    int n0 = nt * 16;
    float bval = b1[n0 + nn];
    v8f d;
#pragma unroll
    for (int r = 0; r < 8; ++r) d[r] = bval;
#pragma unroll
    for (int kt = 0; kt < 8; ++kt) {
      v16bf a  = load_a_packed(A1, kt, lane);
      v16bf bb = load_b_packed(W1p, 32, kt, nt, lane);
      d = wmma_bf16(a, bb, d);
    }
    if (PASS == 1) {
      float s = 0.f, q = 0.f;
#pragma unroll
      for (int r = 0; r < 8; ++r) {
        float mk = maskS[wv][mb + r];
        s += mk * d[r];
        q += mk * d[r] * d[r];
      }
      s += __shfl_down(s, 16, 32);
      q += __shfl_down(q, 16, 32);
      if (lane < 16) {
        atomicAdd(&stats[n0 + nn], s);
        atomicAdd(&stats[512 + n0 + nn], q);
      }
    } else {
      float cnt  = stats[1280];
      float mean = stats[n0 + nn] / cnt;
      float var  = stats[512 + n0 + nn] / cnt - mean * mean;
      float inv  = rsqrtf(var + 1e-5f);
#pragma unroll
      for (int r = 0; r < 8; ++r) {
        float hv = fmaxf((d[r] - mean) * inv, 0.f);
        A2[afrag_idx(mb + r, n0 + nn)] = (__bf16)hv;
      }
    }
  }
  if (PASS == 1) return;
  __syncthreads();

  // ---- layer 2: [16,512] @ [512,128] + b2 ----
  for (int nt = 0; nt < 8; ++nt) {
    int n0 = nt * 16;
    float bval = b2[n0 + nn];
    v8f d;
#pragma unroll
    for (int r = 0; r < 8; ++r) d[r] = bval;
#pragma unroll
    for (int kt = 0; kt < 16; ++kt) {
      v16bf a  = load_a_packed(A2, kt, lane);
      v16bf bb = load_b_packed(W2p, 8, kt, nt, lane);
      d = wmma_bf16(a, bb, d);
    }
    if (PASS == 2) {
      float s = 0.f, q = 0.f;
#pragma unroll
      for (int r = 0; r < 8; ++r) {
        float mk = maskS[wv][mb + r];
        s += mk * d[r];
        q += mk * d[r] * d[r];
      }
      s += __shfl_down(s, 16, 32);
      q += __shfl_down(q, 16, 32);
      if (lane < 16) {
        atomicAdd(&stats[1024 + n0 + nn], s);
        atomicAdd(&stats[1152 + n0 + nn], q);
      }
    } else {
      float cnt  = stats[1280];
      float mean = stats[1024 + n0 + nn] / cnt;
      float var  = stats[1152 + n0 + nn] / cnt - mean * mean;
      float inv  = rsqrtf(var + 1e-5f);
#pragma unroll
      for (int r = 0; r < 8; ++r)
        H2[(mb + r) * 128 + n0 + nn] = fmaxf((d[r] - mean) * inv, 0.f);
    }
  }
  if (PASS == 2) return;
  __syncthreads();

  // ---- layer 3 + sigmoid + self-loop override ----
  if (lane < 16) {
    int e = e0 + lane;
    float acc = b3[0];
    for (int c = 0; c < 128; ++c) acc += H2[lane * 128 + c] * W3[c];
    float a_ = (maskS[wv][lane] == 0.f) ? 1.f : (1.f / (1.f + __expf(-acc)));
    att[e] = a_;
  }
}

// ---------------- reverse-edge averaging (closed-form perm) -------------------
__global__ void k_pair(const float* __restrict__ att, float* __restrict__ out,
                       int E, int EH) {
  int e = blockIdx.x * blockDim.x + threadIdx.x;
  if (e >= E) return;
  int rev = (e < EH) ? e + EH : (e < 2 * EH) ? e - EH : e;
  out[e] = 0.5f * (att[e] + att[rev]);
}

extern "C" void kernel_launch(void* const* d_in, const int* in_sizes, int n_in,
                              void* d_out, int out_size, void* d_ws, size_t ws_size,
                              hipStream_t stream) {
  const float* h_in = (const float*)d_in[0];
  const int* edge_index = (const int*)d_in[1];
  const float* Wq = (const float*)d_in[2];
  const float* bq = (const float*)d_in[3];
  const float* Wk = (const float*)d_in[4];
  const float* bk = (const float*)d_in[5];
  const float* Wv = (const float*)d_in[6];
  const float* bv = (const float*)d_in[7];
  const float* W1 = (const float*)d_in[8];
  const float* b1 = (const float*)d_in[9];
  const float* W2 = (const float*)d_in[10];
  const float* b2 = (const float*)d_in[11];
  const float* W3 = (const float*)d_in[12];
  const float* b3 = (const float*)d_in[13];
  float* out = (float*)d_out;

  int E  = in_sizes[1] / 2;
  int EH = (E - NNODE) / 2;
  const int* row = edge_index;
  const int* col = edge_index + E;

  // workspace layout (all offsets 32B-aligned)
  float* emb   = (float*)d_ws;          // 4096*128
  float* stats = emb + NNODE * CCH;     // 1536 floats (1281 used)
  float* att   = stats + 1536;          // E floats
  __bf16* Wqp = (__bf16*)(att + E);
  __bf16* Wkp = Wqp + 128 * 128;
  __bf16* Wvp = Wkp + 128 * 128;
  __bf16* W1p = Wvp + 128 * 128;        // 256x512
  __bf16* W2p = W1p + 256 * 512;        // 512x128

  int nz = NNODE * CCH + 1536;
  k_zero<<<dim3((nz + 255) / 256), dim3(256), 0, stream>>>(emb, nz);
  k_pack<<<dim3(64), dim3(256), 0, stream>>>(Wq, Wqp, 128, 128);
  k_pack<<<dim3(64), dim3(256), 0, stream>>>(Wk, Wkp, 128, 128);
  k_pack<<<dim3(64), dim3(256), 0, stream>>>(Wv, Wvp, 128, 128);
  k_pack<<<dim3(512), dim3(256), 0, stream>>>(W1, W1p, 256, 512);
  k_pack<<<dim3(256), dim3(256), 0, stream>>>(W2, W2p, 512, 128);

  k_attn<<<dim3(8 * NNODE), dim3(128), 0, stream>>>(h_in, Wqp, Wkp, Wvp, bq, bk, bv, emb);

  int mlp_blocks = E / 32;  // 2 waves x 16 edges per block
  k_mlp<1><<<dim3(mlp_blocks), dim3(64), 0, stream>>>(emb, row, col, W1p, b1, W2p, b2,
                                                      W3, b3, stats, att, E);
  k_mlp<2><<<dim3(mlp_blocks), dim3(64), 0, stream>>>(emb, row, col, W1p, b1, W2p, b2,
                                                      W3, b3, stats, att, E);
  k_mlp<3><<<dim3(mlp_blocks), dim3(64), 0, stream>>>(emb, row, col, W1p, b1, W2p, b2,
                                                      W3, b3, stats, att, E);

  k_pair<<<dim3((E + 255) / 256), dim3(256), 0, stream>>>(att, out, E, EH);
}
